// gc_gcn_32006096290304
// MI455X (gfx1250) — matrix operand order repaired and checked by reading, not compiled, not run on hardware
//
#include <hip/hip_runtime.h>
#include <hip/hip_bf16.h>
#include <math.h>

// Problem constants (match the JAX reference)
#define BB   64   // batch
#define CC   64   // in channels
#define TT   128  // frames
#define VV   25   // joints
#define RR   8    // rel channels
#define TS   9    // temporal taps
#define OUTC 64   // out channels
#define TVTOT (TT * VV)           // 3200
#define PADV  32                  // V padded to WMMA N/K granularity

typedef __attribute__((ext_vector_type(16))) _Float16 v16h;
typedef __attribute__((ext_vector_type(8)))  _Float16 v8h;
typedef __attribute__((ext_vector_type(8)))  float    v8f;

union AFrag { v16h v; v8h h[2]; };

// ---------------------------------------------------------------------------
// Kernel 1: xm = mean_t(x);  x1 = W1@xm + b1;  x2 = W2@xm + b2;
//           rel[b,r,v,i] = tanh(x1[r,v] - x2[r,i])   (fp32, to workspace)
// ---------------------------------------------------------------------------
__global__ __launch_bounds__(256)
void k1_rel(const float* __restrict__ x, const float* __restrict__ W1,
            const float* __restrict__ b1, const float* __restrict__ W2,
            const float* __restrict__ b2, float* __restrict__ rel)
{
    const int b   = blockIdx.x;
    const int tid = threadIdx.x;
    __shared__ float xm[CC * VV];          // time-mean of x, per (c,v)
    __shared__ float x1s[RR * VV];
    __shared__ float x2s[RR * VV];

    const float* xb = x + (size_t)b * CC * TT * VV;
    for (int idx = tid; idx < CC * VV; idx += 256) {
        const int c = idx / VV, v = idx % VV;
        const float* p = xb + (size_t)c * TT * VV + v;
        float s = 0.f;
        #pragma unroll 4
        for (int t = 0; t < TT; ++t) s += p[t * VV];
        xm[idx] = s * (1.0f / (float)TT);  // mean over time
    }
    __syncthreads();

    for (int idx = tid; idx < 2 * RR * VV; idx += 256) {
        const int which = idx / (RR * VV);   // 0 -> x1, 1 -> x2
        const int rv = idx % (RR * VV);
        const int r = rv / VV, v = rv % VV;
        const float* W = which ? W2 : W1;
        float s = which ? b2[r] : b1[r];
        #pragma unroll 8
        for (int c = 0; c < CC; ++c) s += W[r * CC + c] * xm[c * VV + v];
        (which ? x2s : x1s)[rv] = s;
    }
    __syncthreads();

    float* relb = rel + (size_t)b * RR * VV * VV;
    for (int idx = tid; idx < RR * VV * VV; idx += 256) {
        const int r = idx / (VV * VV);
        const int rem = idx % (VV * VV);
        const int v = rem / VV, i = rem % VV;
        relb[idx] = tanhf(x1s[r * VV + v] - x2s[r * VV + i]);
    }
}

// ---------------------------------------------------------------------------
// Kernel 2 (dominant, WMMA): per (b,c) block:
//   M_k[v,i] = sum_r W4[c*9+k,r]*rel[b,r,v,i] + b4[c*9+k] + A[v,i]
//   z[t,i]   = sum_k sum_v x[b,c,t+k-8,v] * M_k[v,i]
// Tiled as 9 GEMMs of (128x32)x(32x32) in f16 WMMA, f32 accumulate.
// z written as f16 to zt[b][t*25+i][c] (c contiguous for conv3's K dim).
// ---------------------------------------------------------------------------
__global__ __launch_bounds__(128)
void k2_tgconv(const float* __restrict__ x, const float* __restrict__ A,
               const float* __restrict__ W4, const float* __restrict__ b4,
               const float* __restrict__ rel, _Float16* __restrict__ zt)
{
    const int bc = blockIdx.x;
    const int b = bc >> 6, c = bc & 63;
    const int tid = threadIdx.x;

    __shared__ __align__(32) _Float16 xs[TT + TS - 1][PADV];   // 136x32 f16 (causal pad rows 0..7)
    __shared__ __align__(32) _Float16 mt[TS][PADV][PADV];      // mt[k][i][v] = M_k[v][i]
    __shared__ float rels[RR][VV][VV];
    __shared__ float As[VV][VV];
    __shared__ float w4s[TS][RR];
    __shared__ float b4s[TS];

    // ---- stage rel[b], A, W4 rows for this c, b4 ----
    const float* relb = rel + (size_t)b * RR * VV * VV;
    for (int idx = tid; idx < RR * VV * VV; idx += 128) (&rels[0][0][0])[idx] = relb[idx];
    for (int idx = tid; idx < VV * VV; idx += 128)      (&As[0][0])[idx]      = A[idx];
    if (tid < TS * RR) (&w4s[0][0])[tid] = W4[(size_t)(c * TS) * RR + tid];
    if (tid < TS)      b4s[tid] = b4[c * TS + tid];

    // ---- stage x slab (f16, causal left-pad of TS-1 rows, cols padded) ----
    const float* xb = x + ((size_t)b * CC + c) * TT * VV;
    for (int idx = tid; idx < (TT + TS - 1) * PADV; idx += 128) {
        const int row = idx >> 5, v = idx & 31;
        float val = 0.f;
        if (row >= TS - 1 && v < VV) val = xb[(size_t)(row - (TS - 1)) * VV + v];
        xs[row][v] = (_Float16)val;
    }
    __syncthreads();

    // ---- build transposed tap matrices mt[k][i][v] ----
    for (int idx = tid; idx < TS * PADV * PADV; idx += 128) {
        const int k = idx >> 10;
        const int rem = idx & 1023;
        const int i = rem >> 5, v = rem & 31;
        float s = 0.f;
        if (i < VV && v < VV) {
            s = b4s[k] + As[v][i];
            #pragma unroll
            for (int r = 0; r < RR; ++r) s += w4s[k][r] * rels[r][v][i];
        }
        mt[k][i][v] = (_Float16)s;
    }
    __syncthreads();

    // ---- WMMA: 4 waves, each owns 2 M-tiles (t) x 2 N-tiles (i) ----
    const int lane = tid & 31, w = tid >> 5;
    const int l16 = lane & 15, half = lane >> 4;

    v8f acc[2][2] = {};
    #pragma unroll
    for (int k = 0; k < TS; ++k) {
        AFrag a[2];
        v16h bf[2];
        #pragma unroll
        for (int mi = 0; mi < 2; ++mi) {
            const int row = (2 * w + mi) * 16 + l16 + k;   // <= 135
            // 16-bit A layout: per-lane contiguous 16 K at 16*half
            a[mi].h[0] = *(const v8h*)&xs[row][16 * half];
            a[mi].h[1] = *(const v8h*)&xs[row][16 * half + 8];
        }
        #pragma unroll
        for (int ni = 0; ni < 2; ++ni) {
            // 16-bit B layout: lane = column i, contiguous 16 K at 16*half
            bf[ni] = *(const v16h*)&mt[k][ni * 16 + l16][16 * half];
        }
        #pragma unroll
        for (int mi = 0; mi < 2; ++mi)
            #pragma unroll
            for (int ni = 0; ni < 2; ++ni)
                acc[mi][ni] = __builtin_amdgcn_wmma_f32_16x16x32_f16(
                    false, a[mi].v, false, bf[ni], (short)0, acc[mi][ni],
                    false, false);
    }

    // ---- scatter z tiles -> zt[b][t*25+i][c] (f16) ----
    #pragma unroll
    for (int mi = 0; mi < 2; ++mi) {
        #pragma unroll
        for (int ni = 0; ni < 2; ++ni) {
            const int i = ni * 16 + l16;
            if (i < VV) {
                #pragma unroll
                for (int vg = 0; vg < 8; ++vg) {
                    const int t = (2 * w + mi) * 16 + vg + 8 * half;
                    const size_t tv = (size_t)t * VV + i;
                    zt[((size_t)b * TVTOT + tv) * CC + c] = (_Float16)acc[mi][ni][vg];
                }
            }
        }
    }
}

// ---------------------------------------------------------------------------
// Kernel 3 (WMMA): out[b,o,tv] = sum_c W3[o,c]*z[b,c,tv] + b3[o]
// GEMM per b: D[tv,o] = zt[tv,:] @ W3^T, M=3200, N=64, K=64.
// ---------------------------------------------------------------------------
__global__ __launch_bounds__(256)
void k3_conv3(const _Float16* __restrict__ zt, const float* __restrict__ W3,
              const float* __restrict__ b3, float* __restrict__ out)
{
    const int bid = blockIdx.x;
    const int b = bid / 25, mg = bid % 25;   // 25 blocks per batch, 8 waves each
    const int tid = threadIdx.x;

    __shared__ __align__(32) _Float16 w3s[OUTC][CC];   // row o, c contiguous (= B layout)
    for (int idx = tid; idx < OUTC * CC; idx += 256)
        (&w3s[0][0])[idx] = (_Float16)W3[idx];
    __syncthreads();

    const int lane = tid & 31, w = tid >> 5;
    const int l16 = lane & 15, half = lane >> 4;
    const int mtile = mg * 8 + w;                       // 0..199
    const _Float16* ztb = zt + (size_t)b * TVTOT * CC;

    v8f acc[4] = {};
    #pragma unroll
    for (int kc = 0; kc < 2; ++kc) {                    // K chunks of 32 over c
        AFrag a;
        const _Float16* arow = ztb + (size_t)(mtile * 16 + l16) * CC + kc * 32;
        a.h[0] = *(const v8h*)(arow + 16 * half);
        a.h[1] = *(const v8h*)(arow + 16 * half + 8);
        #pragma unroll
        for (int ni = 0; ni < 4; ++ni) {
            const v16h bf = *(const v16h*)&w3s[ni * 16 + l16][kc * 32 + 16 * half];
            acc[ni] = __builtin_amdgcn_wmma_f32_16x16x32_f16(
                false, a.v, false, bf, (short)0, acc[ni], false, false);
        }
    }

    #pragma unroll
    for (int ni = 0; ni < 4; ++ni) {
        const int o = ni * 16 + l16;
        const float bias = b3[o];
        float* op = out + ((size_t)b * OUTC + o) * TVTOT + mtile * 16 + 8 * half;
        #pragma unroll
        for (int vg = 0; vg < 8; ++vg) op[vg] = acc[ni][vg] + bias;   // contiguous f32
    }
}

// ---------------------------------------------------------------------------
// Launch
// ---------------------------------------------------------------------------
extern "C" void kernel_launch(void* const* d_in, const int* in_sizes, int n_in,
                              void* d_out, int out_size, void* d_ws, size_t ws_size,
                              hipStream_t stream)
{
    // setup_inputs order: x, A, W1, b1, W2, b2, W4, b4, W3, b3
    const float* x  = (const float*)d_in[0];
    const float* A  = (const float*)d_in[1];
    const float* W1 = (const float*)d_in[2];
    const float* b1 = (const float*)d_in[3];
    const float* W2 = (const float*)d_in[4];
    const float* b2 = (const float*)d_in[5];
    const float* W4 = (const float*)d_in[6];
    const float* b4 = (const float*)d_in[7];
    const float* W3 = (const float*)d_in[8];
    const float* b3 = (const float*)d_in[9];
    float* out = (float*)d_out;

    // Workspace layout (fully overwritten every call):
    //   rel : B*R*V*V fp32 = 1,280,000 B   (256-aligned region start)
    //   zt  : B*T*V*C f16  = 26,214,400 B
    const size_t rel_bytes = (size_t)BB * RR * VV * VV * sizeof(float);
    float*    rel_ws = (float*)d_ws;
    _Float16* zt_ws  = (_Float16*)((char*)d_ws + rel_bytes);
    (void)ws_size; (void)in_sizes; (void)n_in; (void)out_size;

    k1_rel<<<BB, 256, 0, stream>>>(x, W1, b1, W2, b2, rel_ws);
    k2_tgconv<<<BB * CC, 128, 0, stream>>>(x, A, W4, b4, rel_ws, zt_ws);
    k3_conv3<<<BB * 25, 256, 0, stream>>>(zt_ws, W3, b3, out);
}
